// DecoderRNN_55018531062036
// MI455X (gfx1250) — matrix operand order repaired and checked by reading, not compiled, not run on hardware
//
#include <hip/hip_runtime.h>
#include <math.h>
#include <stdint.h>

// ---------------------------------------------------------------------------
// DecoderRNN (2-layer GRU, teacher forcing) for MI455X / gfx1250, wave32.
//   V=32000 E=H=1024 L=2 B=32 T=128 (127 steps)
// Strategy:
//   - recurrent GRU: persistent 64-WG kernel, bf16 WMMA 16x16x32, global
//     spin barrier between phases (weights bf16 => 25MB, L2-resident)
//   - output projection factored into one [4064,1024]x[1024,32000] bf16 WMMA
//     GEMM writing logits to d_out (tiles staged via GLOBAL_LOAD_ASYNC_TO_LDS,
//     double-buffered, ASYNCcnt-tracked), then per-row log-softmax.
// Workspace: ~100 MB (bf16 weights + hidden history).
// ---------------------------------------------------------------------------

typedef unsigned short ushort_t;
typedef __attribute__((ext_vector_type(16))) __bf16 v16bf;
typedef __attribute__((ext_vector_type(8)))  float  v8f;

#define VOC   32000
#define EMB   1024
#define HID   1024
#define H3    3072
#define BAT   32
#define TT    128
#define STEPS 127
#define ROWS  (STEPS * BAT)   // 4064
#define NWG   64

union FragU {
    v16bf     v;
    uint4     q[2];
    ushort_t  u[16];
};

__device__ __forceinline__ ushort_t f2bf(float f) {
    unsigned u = __float_as_uint(f);
    u += 0x7FFFu + ((u >> 16) & 1u);   // round-to-nearest-even
    return (ushort_t)(u >> 16);
}

__device__ __forceinline__ v8f wmma_bf16(v16bf a, v16bf b, v8f c) {
    // v_wmma_f32_16x16x32_bf16 : D = A(16x32) * B(32x16) + C(16x16 f32)
    return __builtin_amdgcn_wmma_f32_16x16x32_bf16(
        false, a, false, b, (short)0, c, false, false);
}

// A-matrix frag (16-bit 16x32, ISA 7.12.2): lane<16 -> M=lane, K = {0..7,16..23};
// lane>=16 -> M=lane-16, K = {8..15,24..31}.
__device__ __forceinline__ v16bf load_a(const ushort_t* A, int stride, int m,
                                        int k, int hi) {
    FragU f;
    const ushort_t* p = A + m * stride + k + hi * 8;
    f.q[0] = *(const uint4*)(p);
    f.q[1] = *(const uint4*)(p + 16);
    return f.v;
}

// B-matrix frag (32x16): lane l holds column N = n0 + l%16; K = {0..7,16..23} (l<16)
// or {8..15,24..31} (l>=16) across elements, matching the A frag.
__device__ __forceinline__ v16bf load_b(const ushort_t* Bm, int stride, int k,
                                        int lane, int n0) {
    FragU f;
    const ushort_t* p = Bm + (k + (lane >> 4) * 8) * stride + n0 + (lane & 15);
    #pragma unroll
    for (int s = 0; s < 8; ++s) {
        f.u[s]     = p[s * stride];
        f.u[8 + s] = p[(16 + s) * stride];
    }
    return f.v;
}

// ---- CDNA5 async global->LDS copies (ASYNCcnt-tracked, ISA ch.10/15.18) ----
// Low 32 bits of a flat pointer into the LDS aperture are the LDS byte offset
// (ISA 10.2).
__device__ __forceinline__ void async_copy_b128(void* lds, const void* gaddr) {
    unsigned l = (unsigned)(uintptr_t)lds;
    asm volatile("global_load_async_to_lds_b128 %0, %1, off"
                 :: "v"(l), "v"(gaddr) : "memory");
}
__device__ __forceinline__ void async_copy_b64(void* lds, const void* gaddr) {
    unsigned l = (unsigned)(uintptr_t)lds;
    asm volatile("global_load_async_to_lds_b64 %0, %1, off"
                 :: "v"(l), "v"(gaddr) : "memory");
}
__device__ __forceinline__ void wait_async0() {
    asm volatile("s_wait_asynccnt 0x0" ::: "memory");
}

// ---------------------------------------------------------------------------
// Kernel 1: convert weights to bf16, transposed for WMMA B-matrix access.
//   wIhT/wHhT : [L][E][3H]   (element (l,e,n) = W[l][n][e])
//   outWT     : [H][V]       (element (h,v)   = out_W[v][h])
// dst-linear indexing -> coalesced writes; strided reads amortize in L2.
// ---------------------------------------------------------------------------
__global__ void __launch_bounds__(256) k_convert(
    const float* __restrict__ W_ih, const float* __restrict__ W_hh,
    const float* __restrict__ outW,
    ushort_t* __restrict__ wIhT, ushort_t* __restrict__ wHhT,
    ushort_t* __restrict__ outWT)
{
    const long SW = 2L * HID * H3;        // per-tensor count
    const long SO = (long)VOC * HID;
    const long stride = (long)gridDim.x * blockDim.x;
    const long t0 = (long)blockIdx.x * blockDim.x + threadIdx.x;
    for (long d = t0; d < SW; d += stride) {
        long l = d / ((long)HID * H3);
        long r = d % ((long)HID * H3);
        long e = r / H3, n = r % H3;
        long s = l * (long)H3 * HID + n * HID + e;
        wIhT[d] = f2bf(W_ih[s]);
        wHhT[d] = f2bf(W_hh[s]);
    }
    for (long d = t0; d < SO; d += stride) {
        long h = d / VOC, v = d % VOC;
        outWT[d] = f2bf(outW[v * HID + h]);
    }
}

// ---------------------------------------------------------------------------
// Kernel 2: initial state (h from encoder_hidden, embed token 0, barrier=0)
// ---------------------------------------------------------------------------
__global__ void __launch_bounds__(256) k_init(
    const int* __restrict__ target, const float* __restrict__ enc,
    const float* __restrict__ embW,
    float* __restrict__ h0f, float* __restrict__ h1f,
    ushort_t* __restrict__ h0bf, ushort_t* __restrict__ hist,
    ushort_t* __restrict__ xbf, unsigned* __restrict__ bar)
{
    const int i = blockIdx.x * 256 + threadIdx.x;   // grid covers BAT*HID
    if (i == 0) { bar[0] = 0u; bar[1] = 0u; }
    if (i < BAT * HID) {
        float e0 = enc[i];
        float e1 = enc[BAT * HID + i];
        h0f[i] = e0;
        h1f[i] = e1;
        h0bf[BAT * HID + i] = f2bf(e0);   // parity buffer 1 (read at t=0)
        hist[i] = f2bf(e1);               // hist[0] = initial layer-1 h
        int b = i >> 10, c = i & 1023;
        int tok = target[b * TT + 0];
        xbf[i] = f2bf(embW[(long)tok * EMB + c]);
    }
}

// ---------------------------------------------------------------------------
// Kernel 3: persistent GRU. 64 WGs x 256 thr. WG j owns H-columns [16j,16j+16).
// Waves 0..5: gate matrices (gi/gh x r/z/n) via bf16 WMMA over K=1024.
// Global spin barrier (agent-scope atomics) twice per step.
// ---------------------------------------------------------------------------
__global__ void __launch_bounds__(256) k_gru(
    const int* __restrict__ target, const float* __restrict__ embW,
    const float* __restrict__ b_ih, const float* __restrict__ b_hh,
    const ushort_t* __restrict__ wIhT, const ushort_t* __restrict__ wHhT,
    ushort_t* __restrict__ xbf, ushort_t* __restrict__ h0bf,
    float* __restrict__ h0f, float* __restrict__ h1f,
    ushort_t* __restrict__ hist, unsigned* __restrict__ bar)
{
    __shared__ float gacc[6][32][16];     // [gi_r,gi_z,gi_n,gh_r,gh_z,gh_n]
    const int wg   = blockIdx.x;
    const int tid  = threadIdx.x;
    const int wid  = tid >> 5;
    const int lane = tid & 31;
    const int col0 = wg * 16;
    const int hi   = lane >> 4;
    const int mlo  = lane & 15;
    unsigned gen = 0;

    auto gridbar = [&]() {
        __syncthreads();
        gen++;
        if (tid == 0) {
            unsigned prev = __hip_atomic_fetch_add(&bar[0], 1u,
                __ATOMIC_ACQ_REL, __HIP_MEMORY_SCOPE_AGENT);
            if (prev == gen * NWG - 1u) {
                __hip_atomic_store(&bar[1], gen,
                    __ATOMIC_RELEASE, __HIP_MEMORY_SCOPE_AGENT);
            } else {
                while (__hip_atomic_load(&bar[1],
                        __ATOMIC_ACQUIRE, __HIP_MEMORY_SCOPE_AGENT) < gen)
                    __builtin_amdgcn_s_sleep(8);
            }
        }
        __syncthreads();
    };

    auto gru_layer = [&](const ushort_t* Agi, const ushort_t* Agh,
                         const ushort_t* Wi, const ushort_t* Wh,
                         const float* bi, const float* bh,
                         float* hf, ushort_t* ho1, ushort_t* ho2) {
        if (wid < 6) {
            const int g = wid % 3;                    // r/z/n
            const ushort_t* A = (wid < 3) ? Agi : Agh;
            const ushort_t* W = (wid < 3) ? Wi : Wh;
            const int nbase = g * HID + col0;
            v8f acc0 = {0.f,0.f,0.f,0.f,0.f,0.f,0.f,0.f};
            v8f acc1 = acc0;
            for (int k = 0; k < HID; k += 32) {
                v16bf bf_ = load_b(W, H3, k, lane, nbase);
                v16bf a0  = load_a(A, HID, mlo,      k, hi);
                v16bf a1  = load_a(A, HID, mlo + 16, k, hi);
                acc0 = wmma_bf16(a0, bf_, acc0);
                acc1 = wmma_bf16(a1, bf_, acc1);
            }
            const int n  = lane & 15;
            const int mb = (lane < 16) ? 0 : 8;       // C/D layout (ISA 7.12.2)
            #pragma unroll
            for (int r = 0; r < 8; ++r) {
                gacc[wid][mb + r][n]      = acc0[r];
                gacc[wid][16 + mb + r][n] = acc1[r];
            }
        }
        __syncthreads();
        #pragma unroll
        for (int e = tid; e < BAT * 16; e += 256) {
            const int b = e >> 4, c = e & 15;
            const int col = col0 + c;
            float ir = gacc[0][b][c] + bi[col];
            float iz = gacc[1][b][c] + bi[HID + col];
            float in_ = gacc[2][b][c] + bi[2 * HID + col];
            float hr = gacc[3][b][c] + bh[col];
            float hz = gacc[4][b][c] + bh[HID + col];
            float hn = gacc[5][b][c] + bh[2 * HID + col];
            float rg = 1.f / (1.f + __expf(-(ir + hr)));
            float zg = 1.f / (1.f + __expf(-(iz + hz)));
            float ng = tanhf(in_ + rg * hn);
            float hp = hf[b * HID + col];
            float hv = (1.f - zg) * ng + zg * hp;
            hf[b * HID + col] = hv;
            ushort_t bv = f2bf(hv);
            if (ho1) ho1[b * HID + col] = bv;
            if (ho2) ho2[b * HID + col] = bv;
        }
    };

    for (int t = 0; t < STEPS; ++t) {
        const int cur = t & 1, prv = cur ^ 1;
        // ---- phase A: layer 0 (uses x(t), h0(t-1)) ----
        gru_layer(xbf, h0bf + prv * (BAT * HID),
                  wIhT, wHhT, b_ih, b_hh,
                  h0f, h0bf + cur * (BAT * HID), (ushort_t*)nullptr);
        gridbar();
        // ---- phase B: layer 1 (uses h0(t), h1(t-1)=hist[t]) + embed(t+1) ----
        gru_layer(h0bf + cur * (BAT * HID), hist + (size_t)t * BAT * HID,
                  wIhT + (size_t)HID * H3, wHhT + (size_t)HID * H3,
                  b_ih + H3, b_hh + H3,
                  h1f, hist + (size_t)(t + 1) * BAT * HID, (ushort_t*)nullptr);
        if (t + 1 < STEPS) {
            for (int e = wg * 256 + tid; e < BAT * EMB; e += NWG * 256) {
                int b = e >> 10, c = e & 1023;
                int tok = target[b * TT + (t + 1)];
                xbf[e] = f2bf(embW[(long)tok * EMB + c]);
            }
        }
        gridbar();
    }
}

// ---------------------------------------------------------------------------
// Kernel 4: logits GEMM. [4064,1024](bf16) x [1024,32000](bf16) + out_b.
// WG = 256 thr = 8 waves, 32x128 tile. A/B tiles double-buffered in LDS via
// GLOBAL_LOAD_ASYNC_TO_LDS (ASYNCcnt): issue next tile, WMMA current tile,
// s_wait_asynccnt 0 + barrier, swap. global_prefetch 3 tiles ahead.
// ---------------------------------------------------------------------------
__global__ void __launch_bounds__(256) k_gemm(
    const ushort_t* __restrict__ A,    // [ROWS][1024]
    const ushort_t* __restrict__ Bw,   // [1024][VOC]
    const float* __restrict__ outb, float* __restrict__ out)
{
    __shared__ __align__(16) ushort_t As[2][32][32];
    __shared__ __align__(16) ushort_t Bs[2][32][128];
    const int tid  = threadIdx.x;
    const int wid  = tid >> 5;
    const int lane = tid & 31;
    const int M0 = blockIdx.y * 32;
    const int N0 = blockIdx.x * 128;
    const int hi  = lane >> 4, mlo = lane & 15;
    const int nt0 = wid * 16;
    const int r8  = tid >> 3, c8 = tid & 7;   // 32 rows x 8 chunks

    auto loadTilesAsync = [&](int kt, int buf) {
        const int k = kt * 32;
        // A tile: 32x32 halves; 8B (b64) per thread, async into LDS
        async_copy_b64(&As[buf][r8][c8 * 4],
                       A + (size_t)(M0 + r8) * HID + k + c8 * 4);
        // B tile: 32x128 halves; 2x16B (b128) per thread, async into LDS
        const ushort_t* src = Bw + (size_t)(k + r8) * VOC + N0 + c8 * 16;
        async_copy_b128(&Bs[buf][r8][c8 * 16],     src);
        async_copy_b128(&Bs[buf][r8][c8 * 16 + 8], src + 8);
    };

    v8f acc0 = {0.f,0.f,0.f,0.f,0.f,0.f,0.f,0.f};
    v8f acc1 = acc0;
    loadTilesAsync(0, 0);
    int buf = 0;
    wait_async0();
    __syncthreads();
    for (int kt = 0; kt < 32; ++kt) {
        if (kt + 3 < 32) {  // L2 prefetch for the B tile 3 iterations ahead
            __builtin_prefetch(
                Bw + (size_t)((kt + 3) * 32 + r8) * VOC + N0 + c8 * 16, 0, 1);
        }
        if (kt + 1 < 32) loadTilesAsync(kt + 1, buf ^ 1);  // overlaps WMMA below
        v16bf bf_ = load_b(&Bs[buf][0][0], 128, 0, lane, nt0);
        v16bf a0  = load_a(&As[buf][0][0], 32, mlo,      0, hi);
        v16bf a1  = load_a(&As[buf][0][0], 32, mlo + 16, 0, hi);
        acc0 = wmma_bf16(a0, bf_, acc0);
        acc1 = wmma_bf16(a1, bf_, acc1);
        wait_async0();       // own wave's async loads for next buffer complete
        __syncthreads();     // all waves' tiles complete before swap
        buf ^= 1;
    }
    const int n  = N0 + nt0 + (lane & 15);
    const float bv = outb[n];
    const int mb = (lane < 16) ? 0 : 8;
    #pragma unroll
    for (int r = 0; r < 8; ++r) {
        out[(size_t)(M0 + mb + r) * VOC + n]      = acc0[r] + bv;
        out[(size_t)(M0 + 16 + mb + r) * VOC + n] = acc1[r] + bv;
    }
}

// ---------------------------------------------------------------------------
// Kernel 5: in-place log-softmax per row (max / sumexp / subtract lse).
// ---------------------------------------------------------------------------
__global__ void __launch_bounds__(256) k_lsm(float* __restrict__ out)
{
    __shared__ float red[256];
    float* row = out + (size_t)blockIdx.x * VOC;
    const int tid = threadIdx.x;
    float m = -3.402823466e38f;
    for (int c = tid; c < VOC; c += 256) m = fmaxf(m, row[c]);
    red[tid] = m; __syncthreads();
    for (int s = 128; s; s >>= 1) {
        if (tid < s) red[tid] = fmaxf(red[tid], red[tid + s]);
        __syncthreads();
    }
    m = red[0]; __syncthreads();
    float sum = 0.f;
    for (int c = tid; c < VOC; c += 256) sum += __expf(row[c] - m);
    red[tid] = sum; __syncthreads();
    for (int s = 128; s; s >>= 1) {
        if (tid < s) red[tid] += red[tid + s];
        __syncthreads();
    }
    const float lse = m + __logf(red[0]);
    for (int c = tid; c < VOC; c += 256) row[c] -= lse;
}

// ---------------------------------------------------------------------------
extern "C" void kernel_launch(void* const* d_in, const int* in_sizes, int n_in,
                              void* d_out, int out_size, void* d_ws,
                              size_t ws_size, hipStream_t stream)
{
    const int*   target = (const int*)d_in[0];
    const float* enc    = (const float*)d_in[1];
    const float* embW   = (const float*)d_in[2];
    const float* W_ih   = (const float*)d_in[3];
    const float* W_hh   = (const float*)d_in[4];
    const float* b_ih   = (const float*)d_in[5];
    const float* b_hh   = (const float*)d_in[6];
    const float* outW   = (const float*)d_in[7];
    const float* outb   = (const float*)d_in[8];
    float* out = (float*)d_out;

    // workspace layout (bytes), all offsets 16B aligned; total ~99.5 MB
    char* ws = (char*)d_ws;
    const size_t szW   = 2ull * HID * H3 * 2;        // 12,582,912
    const size_t szOW  = (size_t)HID * VOC * 2;      // 65,536,000
    const size_t szH   = 128ull * BAT * HID * 2;     // 8,388,608
    ushort_t* wIhT  = (ushort_t*)(ws);
    ushort_t* wHhT  = (ushort_t*)(ws + szW);
    ushort_t* outWT = (ushort_t*)(ws + 2 * szW);
    ushort_t* hist  = (ushort_t*)(ws + 2 * szW + szOW);
    ushort_t* xbf   = (ushort_t*)(ws + 2 * szW + szOW + szH);
    ushort_t* h0bf  = (ushort_t*)(ws + 2 * szW + szOW + szH + 65536);
    float*    h0f   = (float*)   (ws + 2 * szW + szOW + szH + 65536 + 131072);
    float*    h1f   = (float*)   (ws + 2 * szW + szOW + szH + 65536 + 131072 + 131072);
    unsigned* bar   = (unsigned*)(ws + 2 * szW + szOW + szH + 65536 + 131072 + 2 * 131072);

    k_convert<<<2048, 256, 0, stream>>>(W_ih, W_hh, outW, wIhT, wHhT, outWT);
    k_init<<<(BAT * HID + 255) / 256, 256, 0, stream>>>(
        target, enc, embW, h0f, h1f, h0bf, hist, xbf, bar);
    k_gru<<<NWG, 256, 0, stream>>>(
        target, embW, b_ih, b_hh, wIhT, wHhT, xbf, h0bf, h0f, h1f, hist, bar);
    // GEMM A = layer-1 hidden history, rows (t,b) -> hist[t+1][b]
    k_gemm<<<dim3(VOC / 128, ROWS / 32), 256, 0, stream>>>(
        hist + (size_t)BAT * HID, outWT, outb, out);
    k_lsm<<<ROWS, 256, 0, stream>>>(out);
    (void)in_sizes; (void)n_in; (void)out_size; (void)ws_size;
}
